// MoEFFN_21285857919578
// MI455X (gfx1250) — matrix operand order repaired and checked by reading, not compiled, not run on hardware
//
#include <hip/hip_runtime.h>

#define DIMD 1024
#define NTOK 8192
#define NEXP 8
#define FDIM 3264

typedef __attribute__((ext_vector_type(16))) __bf16 v16bf;
typedef __attribute__((ext_vector_type(8)))  __bf16 v8bf;
typedef __attribute__((ext_vector_type(4)))  __bf16 v4bf;
typedef __attribute__((ext_vector_type(8)))  float  v8f;

union FragBF { v16bf v; v8bf h[2]; };

// ---- CDNA5 async global->LDS copy (ASYNCcnt path), with safe fallback ----
#if defined(__has_builtin)
#if __has_builtin(__builtin_amdgcn_global_load_async_to_lds_b128)
#define HAVE_ASYNC_LDS 1
#endif
#endif

typedef int v4i __attribute__((ext_vector_type(4)));
typedef __attribute__((address_space(1))) v4i gas_v4i;   // global (prints as __device__)
typedef __attribute__((address_space(3))) v4i lds_v4i;   // LDS (__shared__)

__device__ __forceinline__ void copy16_g2lds(const __bf16* g, __bf16* l) {
#ifdef HAVE_ASYNC_LDS
  __builtin_amdgcn_global_load_async_to_lds_b128((gas_v4i*)g, (lds_v4i*)l, 0, 0);
#else
  *(uint4*)l = *(const uint4*)g;
#endif
}

__device__ __forceinline__ void wait_async_done() {
#ifdef HAVE_ASYNC_LDS
#if __has_builtin(__builtin_amdgcn_s_wait_asynccnt)
  __builtin_amdgcn_s_wait_asynccnt(0);
#else
  asm volatile("s_wait_asynccnt 0" ::: "memory");
#endif
#endif
}

// ---------------- kernel: convert x (f32) -> bf16 ----------------
__global__ void moe_xcvt_kernel(const float* __restrict__ x, __bf16* __restrict__ xb) {
  size_t i = (size_t)(blockIdx.x * blockDim.x + threadIdx.x) * 4;
  float4 v = *(const float4*)(x + i);
  v4bf o;
  o[0] = (__bf16)v.x; o[1] = (__bf16)v.y; o[2] = (__bf16)v.z; o[3] = (__bf16)v.w;
  *(v4bf*)(xb + i) = o;
}

// ---------------- kernel: router (softmax + top-2 + norm) ----------------
__global__ void moe_router_kernel(const float* __restrict__ x,
                                  const float* __restrict__ gw,
                                  const float* __restrict__ gb,
                                  float* __restrict__ comb) {
  int wave = threadIdx.x >> 5;
  int lane = threadIdx.x & 31;
  int t = blockIdx.x * 8 + wave;
  const float* xr = x + (size_t)t * DIMD;

  float acc[NEXP];
#pragma unroll
  for (int e = 0; e < NEXP; ++e) acc[e] = 0.f;

  for (int d = lane; d < DIMD; d += 32) {
    float xv = xr[d];
#pragma unroll
    for (int e = 0; e < NEXP; ++e) acc[e] += xv * gw[e * DIMD + d];
  }
#pragma unroll
  for (int e = 0; e < NEXP; ++e) {
#pragma unroll
    for (int off = 16; off > 0; off >>= 1)
      acc[e] += __shfl_down(acc[e], off, 32);
  }

  if (lane == 0) {
    float lg[NEXP];
    float mx = -1e30f;
#pragma unroll
    for (int e = 0; e < NEXP; ++e) { lg[e] = acc[e] + gb[e]; mx = fmaxf(mx, lg[e]); }
    float s[NEXP];
    float se = 0.f;
#pragma unroll
    for (int e = 0; e < NEXP; ++e) { s[e] = __expf(lg[e] - mx); se += s[e]; }
    float inv_se = 1.f / se;
#pragma unroll
    for (int e = 0; e < NEXP; ++e) s[e] *= inv_se;

    int i0 = 0;
#pragma unroll
    for (int e = 1; e < NEXP; ++e) if (s[e] > s[i0]) i0 = e;
    int i1 = (i0 == 0) ? 1 : 0;
#pragma unroll
    for (int e = 0; e < NEXP; ++e) if (e != i0 && s[e] > s[i1]) i1 = e;

    float w0 = s[i0], w1 = s[i1];
    float inv = 1.f / (w0 + w1 + 1e-20f);
#pragma unroll
    for (int e = 0; e < NEXP; ++e) {
      float w = (e == i0) ? w0 * inv : (e == i1) ? w1 * inv : 0.f;
      comb[(size_t)t * NEXP + e] = w;
    }
  }
}

// ---------------- pass A: H = silu((X@Gp^T+gpb) * (X@Up^T+upb)) ----------------
// tile: 128 tokens x 64 f-cols per WG; 8 waves, each 32x32 (2x2 WMMA) x {G,U};
// K-step 32, double-buffered LDS, async X staging.
__global__ __launch_bounds__(256)
void moe_upgate_kernel(const __bf16* __restrict__ xb,
                       const float* __restrict__ gpw, const float* __restrict__ gpb,
                       const float* __restrict__ upw, const float* __restrict__ upb,
                       __bf16* __restrict__ H) {
  __shared__ __bf16 sX[2][128 * 32];
  __shared__ __bf16 sG[2][64 * 32];
  __shared__ __bf16 sU[2][64 * 32];

  const int tid = threadIdx.x;
  const int wave = tid >> 5, lane = tid & 31;
  const int waveM = wave >> 1, waveN = wave & 1;
  const int mBase = blockIdx.x * 128;
  const int fBase = blockIdx.y * 64;
  const int rA = lane & 15, hi = lane >> 4;

  const v8f zero8 = {0.f, 0.f, 0.f, 0.f, 0.f, 0.f, 0.f, 0.f};
  v8f accG[2][2], accU[2][2];
#pragma unroll
  for (int mt = 0; mt < 2; ++mt)
#pragma unroll
    for (int nt = 0; nt < 2; ++nt) { accG[mt][nt] = zero8; accU[mt][nt] = zero8; }

  auto stage = [&](int kk, int buf) {
    // X tile: 128 rows x 32 bf16 (async, no conversion)
#pragma unroll
    for (int i = 0; i < 2; ++i) {
      int s = tid + i * 256;
      int r = s >> 2, c = (s & 3) * 8;
      copy16_g2lds(xb + (size_t)(mBase + r) * DIMD + kk + c, &sX[buf][r * 32 + c]);
    }
    // Gp/Up tiles: 64 rows x 32 f32 -> bf16
#pragma unroll
    for (int i = 0; i < 2; ++i) {
      int s = tid + i * 256;
      int r = s >> 3, c = (s & 7) * 4;
      const size_t off = (size_t)(fBase + r) * DIMD + kk + c;
      float4 g = *(const float4*)(gpw + off);
      float4 u = *(const float4*)(upw + off);
      v4bf g4, u4;
      g4[0] = (__bf16)g.x; g4[1] = (__bf16)g.y; g4[2] = (__bf16)g.z; g4[3] = (__bf16)g.w;
      u4[0] = (__bf16)u.x; u4[1] = (__bf16)u.y; u4[2] = (__bf16)u.z; u4[3] = (__bf16)u.w;
      *(v4bf*)(&sG[buf][r * 32 + c]) = g4;
      *(v4bf*)(&sU[buf][r * 32 + c]) = u4;
      if (kk + 32 < DIMD) {
        __builtin_prefetch(gpw + off + 32, 0, 1);
        __builtin_prefetch(upw + off + 32, 0, 1);
      }
    }
  };

  stage(0, 0);
  const int NIT = DIMD / 32;
  for (int it = 0; it < NIT; ++it) {
    const int cur = it & 1, nxt = cur ^ 1;
    wait_async_done();
    __syncthreads();
    if (it + 1 < NIT) stage((it + 1) * 32, nxt);

    FragBF a[2], bg[2], bu[2];
#pragma unroll
    for (int mt = 0; mt < 2; ++mt) {
      int row = waveM * 32 + mt * 16 + rA;
      a[mt].h[0] = *(const v8bf*)(&sX[cur][row * 32 + hi * 16]);
      a[mt].h[1] = *(const v8bf*)(&sX[cur][row * 32 + hi * 16 + 8]);
    }
#pragma unroll
    for (int nt = 0; nt < 2; ++nt) {
      int col = waveN * 32 + nt * 16 + rA;
      bg[nt].h[0] = *(const v8bf*)(&sG[cur][col * 32 + hi * 16]);
      bg[nt].h[1] = *(const v8bf*)(&sG[cur][col * 32 + hi * 16 + 8]);
      bu[nt].h[0] = *(const v8bf*)(&sU[cur][col * 32 + hi * 16]);
      bu[nt].h[1] = *(const v8bf*)(&sU[cur][col * 32 + hi * 16 + 8]);
    }
#pragma unroll
    for (int mt = 0; mt < 2; ++mt)
#pragma unroll
      for (int nt = 0; nt < 2; ++nt) {
        accG[mt][nt] = __builtin_amdgcn_wmma_f32_16x16x32_bf16(
            false, a[mt].v, false, bg[nt].v, (short)0, accG[mt][nt], false, false);
        accU[mt][nt] = __builtin_amdgcn_wmma_f32_16x16x32_bf16(
            false, a[mt].v, false, bu[nt].v, (short)0, accU[mt][nt], false, false);
      }
  }

  // epilogue: h = silu(g*u), store bf16
#pragma unroll
  for (int mt = 0; mt < 2; ++mt)
#pragma unroll
    for (int nt = 0; nt < 2; ++nt) {
      int f = fBase + waveN * 32 + nt * 16 + rA;
      float gbv = gpb[f], ubv = upb[f];
#pragma unroll
      for (int r = 0; r < 8; ++r) {
        int row = mBase + waveM * 32 + mt * 16 + r + (hi << 3);
        float g = accG[mt][nt][r] + gbv;
        float u = accU[mt][nt][r] + ubv;
        float z = g * u;
        float hval = z / (1.f + __expf(-z));
        H[(size_t)row * FDIM + f] = (__bf16)hval;
      }
    }
}

// ---------------- pass B: y (+)= comb[:,e] * (H @ DW^T + db) ----------------
__global__ __launch_bounds__(256)
void moe_down_kernel(const __bf16* __restrict__ H,
                     const float* __restrict__ dw, const float* __restrict__ db,
                     const float* __restrict__ comb, int e, int accumulate,
                     float* __restrict__ out) {
  __shared__ __bf16 sH[2][128 * 32];
  __shared__ __bf16 sW[2][64 * 32];

  const int tid = threadIdx.x;
  const int wave = tid >> 5, lane = tid & 31;
  const int waveM = wave >> 1, waveN = wave & 1;
  const int mBase = blockIdx.x * 128;
  const int dBase = blockIdx.y * 64;
  const int rA = lane & 15, hi = lane >> 4;

  const v8f zero8 = {0.f, 0.f, 0.f, 0.f, 0.f, 0.f, 0.f, 0.f};
  v8f acc[2][2];
#pragma unroll
  for (int mt = 0; mt < 2; ++mt)
#pragma unroll
    for (int nt = 0; nt < 2; ++nt) acc[mt][nt] = zero8;

  auto stage = [&](int kk, int buf) {
    // H tile: 128 rows x 32 bf16 (async, no conversion)
#pragma unroll
    for (int i = 0; i < 2; ++i) {
      int s = tid + i * 256;
      int r = s >> 2, c = (s & 3) * 8;
      copy16_g2lds(H + (size_t)(mBase + r) * FDIM + kk + c, &sH[buf][r * 32 + c]);
    }
    // DW tile: 64 rows x 32 f32 -> bf16
#pragma unroll
    for (int i = 0; i < 2; ++i) {
      int s = tid + i * 256;
      int r = s >> 3, c = (s & 7) * 4;
      const size_t off = (size_t)(dBase + r) * FDIM + kk + c;
      float4 w = *(const float4*)(dw + off);
      v4bf w4;
      w4[0] = (__bf16)w.x; w4[1] = (__bf16)w.y; w4[2] = (__bf16)w.z; w4[3] = (__bf16)w.w;
      *(v4bf*)(&sW[buf][r * 32 + c]) = w4;
      if (kk + 32 < FDIM) __builtin_prefetch(dw + off + 32, 0, 1);
    }
  };

  stage(0, 0);
  const int NIT = FDIM / 32;
  for (int it = 0; it < NIT; ++it) {
    const int cur = it & 1, nxt = cur ^ 1;
    wait_async_done();
    __syncthreads();
    if (it + 1 < NIT) stage((it + 1) * 32, nxt);

    FragBF a[2], b[2];
#pragma unroll
    for (int mt = 0; mt < 2; ++mt) {
      int row = waveM * 32 + mt * 16 + rA;
      a[mt].h[0] = *(const v8bf*)(&sH[cur][row * 32 + hi * 16]);
      a[mt].h[1] = *(const v8bf*)(&sH[cur][row * 32 + hi * 16 + 8]);
    }
#pragma unroll
    for (int nt = 0; nt < 2; ++nt) {
      int col = waveN * 32 + nt * 16 + rA;
      b[nt].h[0] = *(const v8bf*)(&sW[cur][col * 32 + hi * 16]);
      b[nt].h[1] = *(const v8bf*)(&sW[cur][col * 32 + hi * 16 + 8]);
    }
#pragma unroll
    for (int mt = 0; mt < 2; ++mt)
#pragma unroll
      for (int nt = 0; nt < 2; ++nt)
        acc[mt][nt] = __builtin_amdgcn_wmma_f32_16x16x32_bf16(
            false, a[mt].v, false, b[nt].v, (short)0, acc[mt][nt], false, false);
  }

#pragma unroll
  for (int mt = 0; mt < 2; ++mt)
#pragma unroll
    for (int nt = 0; nt < 2; ++nt) {
      int d = dBase + waveN * 32 + nt * 16 + rA;
      float dbv = db[d];
#pragma unroll
      for (int r = 0; r < 8; ++r) {
        int row = mBase + waveM * 32 + mt * 16 + r + (hi << 3);
        float w = comb[(size_t)row * NEXP + e];
        float val = w * (acc[mt][nt][r] + dbv);
        size_t idx = (size_t)row * DIMD + d;
        if (accumulate) out[idx] += val;
        else out[idx] = val;
      }
    }
}

// ---------------- host entry ----------------
extern "C" void kernel_launch(void* const* d_in, const int* in_sizes, int n_in,
                              void* d_out, int out_size, void* d_ws, size_t ws_size,
                              hipStream_t stream) {
  const float* x      = (const float*)d_in[0];
  const float* gate_w = (const float*)d_in[1];
  const float* gate_b = (const float*)d_in[2];
  const float* up_w   = (const float*)d_in[3];
  const float* up_b   = (const float*)d_in[4];
  const float* gp_w   = (const float*)d_in[5];
  const float* gp_b   = (const float*)d_in[6];
  const float* down_w = (const float*)d_in[7];
  const float* down_b = (const float*)d_in[8];
  float* out = (float*)d_out;

  char* ws = (char*)d_ws;
  __bf16* xb = (__bf16*)ws;
  size_t off = (size_t)NTOK * DIMD * sizeof(__bf16);      // 16 MB
  float* comb = (float*)(ws + off);
  off += (size_t)NTOK * NEXP * sizeof(float);             // +256 KB
  __bf16* H = (__bf16*)(ws + off);                        // +53.5 MB

  moe_xcvt_kernel<<<(NTOK * DIMD / 4) / 256, 256, 0, stream>>>(x, xb);
  moe_router_kernel<<<NTOK / 8, 256, 0, stream>>>(x, gate_w, gate_b, comb);

  for (int e = 0; e < NEXP; ++e) {
    moe_upgate_kernel<<<dim3(NTOK / 128, FDIM / 64), 256, 0, stream>>>(
        xb,
        gp_w + (size_t)e * FDIM * DIMD, gp_b + (size_t)e * FDIM,
        up_w + (size_t)e * FDIM * DIMD, up_b + (size_t)e * FDIM,
        H);
    moe_down_kernel<<<dim3(NTOK / 128, DIMD / 64), 256, 0, stream>>>(
        H,
        down_w + (size_t)e * DIMD * FDIM, down_b + (size_t)e * DIMD,
        comb, e, e > 0 ? 1 : 0, out);
  }
}
